// DySample_8469675508195
// MI455X (gfx1250) — matrix operand order repaired and checked by reading, not compiled, not run on hardware
//
#include <hip/hip_runtime.h>
#include <math.h>

typedef __attribute__((ext_vector_type(16))) _Float16 v16h;
typedef __attribute__((ext_vector_type(8)))  float    v8f;

#define Bn   32
#define Cc   256
#define Hh   80
#define Ww   80
#define MIDc 64
#define H2c  160
#define W2c  160
#define KTILE 32
#define WAVES 4   // waves per block (128 threads, wave32)

union AFrag { v16h v; _Float16 h[16]; };
union CFrag { v8f  v; float    f[8];  };

// -------- Kernel 1: offset network via WMMA, emits clipped sample coords ----
__global__ __launch_bounds__(128) void dysample_offsets(
    const float* __restrict__ x,  const float* __restrict__ w1,
    const float* __restrict__ w2, const float* __restrict__ b2,
    float* __restrict__ IX, float* __restrict__ IY)
{
    __shared__ _Float16 w1h[MIDc * Cc];           // 32 KB, f16 weights
    __shared__ float    xtile[WAVES][KTILE][16];  // 8 KB, per-wave K-slab
    __shared__ float    hid[WAVES][MIDc][16];     // 16 KB, relu(x·w1T)

    const int tid  = threadIdx.x;
    const int wave = tid >> 5;
    const int lane = tid & 31;
    const int half = lane >> 4;
    const int l16  = lane & 15;

    // Stage w1 into LDS as f16 (once per block)
    for (int i = tid; i < MIDc * Cc; i += 128) w1h[i] = (_Float16)w1[i];
    __syncthreads();

    // Tile: 16 consecutive w positions at fixed (b, h). 32*80*5 = 12800 tiles.
    const int tiles_w = Ww / 16;                      // 5
    const int t  = blockIdx.x * WAVES + wave;
    const int wt = t % tiles_w;
    const int h  = (t / tiles_w) % Hh;
    const int b  = t / (tiles_w * Hh);
    const int w0 = wt * 16;

    const float* xb = x + (size_t)b * Cc * Hh * Ww + (size_t)h * Ww + w0;

    CFrag acc[4] = {};   // 4 M-tiles of 16 rows -> MID=64

    for (int kc = 0; kc < Cc; kc += KTILE) {
        // Cooperative coalesced load: lane -> channel kc+lane, 16 floats of row h
        {
            const float* src = xb + (size_t)(kc + lane) * (Hh * Ww);
            float4 p0 = *(const float4*)(src + 0);
            float4 p1 = *(const float4*)(src + 4);
            float4 p2 = *(const float4*)(src + 8);
            float4 p3 = *(const float4*)(src + 12);
            float* dst = &xtile[wave][lane][0];
            *(float4*)(dst + 0)  = p0;  *(float4*)(dst + 4)  = p1;
            *(float4*)(dst + 8)  = p2;  *(float4*)(dst + 12) = p3;
        }
        __syncthreads();

        // B fragment (32x16, K x N): lanes 0-15 K=0..15, lanes 16-31 K=16..31
        AFrag bf;
        #pragma unroll
        for (int v = 0; v < 8; ++v) {
            int k = half * 16 + v * 2;
            bf.h[2*v]   = (_Float16)xtile[wave][k][l16];
            bf.h[2*v+1] = (_Float16)xtile[wave][k+1][l16];
        }

        #pragma unroll
        for (int mt = 0; mt < 4; ++mt) {
            // A fragment (16x32, M x K): lanes 0-15 K=0..15, lanes 16-31 K=16..31
            AFrag af;
            const _Float16* wrow = &w1h[(mt * 16 + l16) * Cc + kc];
            #pragma unroll
            for (int v = 0; v < 8; ++v) {
                int k = half * 16 + v * 2;
                af.h[2*v]   = wrow[k];
                af.h[2*v+1] = wrow[k+1];
            }
            acc[mt].v = __builtin_amdgcn_wmma_f32_16x16x32_f16(
                false, af.v, false, bf.v, (short)0, acc[mt].v, false, false);
        }
        __syncthreads();
    }

    // relu(acc) -> LDS. D layout: vgpr r => M = mt*16 + half*8 + r, N = l16
    #pragma unroll
    for (int mt = 0; mt < 4; ++mt) {
        #pragma unroll
        for (int r = 0; r < 8; ++r) {
            float v = acc[mt].f[r];
            hid[wave][mt * 16 + half * 8 + r][l16] = v > 0.f ? v : 0.f;
        }
    }
    __syncthreads();

    // Tiny second layer (8x64) + tanh + grid math: lanes 0..15, one position each
    if (half == 0) {
        const int n = l16;
        const int w = w0 + n;
        float off[8];
        #pragma unroll
        for (int o = 0; o < 8; ++o) {
            float s = b2[o];
            const float* w2r = w2 + o * MIDc;
            #pragma unroll 8
            for (int m = 0; m < MIDc; ++m) s += hid[wave][m][n] * w2r[m];
            off[o] = tanhf(s);
        }
        const float sf = 2.0f / (float)W2c;   // 2 / max(H2, W2)
        #pragma unroll
        for (int o = 0; o < 4; ++o) {
            int i = o >> 1, j = o & 1;        // o = i*S + j (y-sub, x-sub)
            float gx = -1.0f + (2.0f * (float)(w * 2 + j) + 1.0f) / (float)W2c;
            float gy = -1.0f + (2.0f * (float)(h * 2 + i) + 1.0f) / (float)H2c;
            float sx = gx + off[o]     * sf;
            float sy = gy + off[o + 4] * sf;
            float ix = fminf(fmaxf((sx + 1.0f) * 0.5f * (float)(Ww - 1), 0.0f), (float)(Ww - 1));
            float iy = fminf(fmaxf((sy + 1.0f) * 0.5f * (float)(Hh - 1), 0.0f), (float)(Hh - 1));
            size_t oidx = (((size_t)b * 4 + o) * Hh + h) * Ww + w;
            IX[oidx] = ix;
            IY[oidx] = iy;
        }
    }
}

// -------- Kernel 2: bandwidth gather/blend, 4 outputs + b128 store/thread ---
__global__ __launch_bounds__(256) void dysample_gather4(
    const float* __restrict__ x, const float* __restrict__ IX,
    const float* __restrict__ IY, float4* __restrict__ out)
{
    // Each thread: 4 consecutive x2 at fixed (b, c, y2) -> one 128-bit store
    size_t qidx = (size_t)blockIdx.x * 256 + threadIdx.x;
    int xq = (int)(qidx % (W2c / 4));          // 40 quads per output row
    size_t r = qidx / (W2c / 4);
    int y2 = (int)(r % H2c); r /= H2c;
    int c  = (int)(r % Cc);
    int b  = (int)(r / Cc);

    const int i = y2 & 1;                      // y subpixel
    const int h = y2 >> 1;
    const float* xb = x + ((size_t)b * Cc + c) * (Hh * Ww);
    // pidx = ((b*4 + (i*2 + j))*Hh + h)*Ww + w
    const size_t pbase = ((size_t)b * 4 + i * 2) * (Hh * Ww) + (size_t)h * Ww;

    float res[4];
    #pragma unroll
    for (int e = 0; e < 4; ++e) {
        int x2 = xq * 4 + e;
        int j  = x2 & 1;
        int w  = x2 >> 1;
        size_t pidx = pbase + (size_t)j * (Hh * Ww) + w;
        float ix = IX[pidx], iy = IY[pidx];

        float fx0 = floorf(ix), fy0 = floorf(iy);
        float wx = ix - fx0,    wy = iy - fy0;
        int x0i = (int)fx0,     y0i = (int)fy0;
        int x1i = min(x0i + 1, Ww - 1);
        int y1i = min(y0i + 1, Hh - 1);

        const float* r0 = xb + (size_t)y0i * Ww;
        const float* r1 = xb + (size_t)y1i * Ww;
        float v00 = r0[x0i], v01 = r0[x1i], v10 = r1[x0i], v11 = r1[x1i];

        res[e] = v00 * (1.f - wx) * (1.f - wy) + v01 * wx * (1.f - wy)
               + v10 * (1.f - wx) * wy         + v11 * wx * wy;
    }
    float4 o4; o4.x = res[0]; o4.y = res[1]; o4.z = res[2]; o4.w = res[3];
    out[qidx] = o4;
}

extern "C" void kernel_launch(void* const* d_in, const int* in_sizes, int n_in,
                              void* d_out, int out_size, void* d_ws, size_t ws_size,
                              hipStream_t stream) {
    const float* x  = (const float*)d_in[0];
    const float* w1 = (const float*)d_in[1];
    const float* w2 = (const float*)d_in[2];
    const float* b2 = (const float*)d_in[3];

    float* IX = (float*)d_ws;
    float* IY = IX + (size_t)Bn * 4 * Hh * Ww;   // 2 * 3.28 MB in workspace

    // 12800 position tiles / 4 waves per block
    dysample_offsets<<<(Bn * Hh * (Ww / 16)) / WAVES, 128, 0, stream>>>(
        x, w1, w2, b2, IX, IY);

    // 32*256*160*160 = 209,715,200 outputs -> 52,428,800 float4 quads
    size_t quads = (size_t)Bn * Cc * H2c * W2c / 4;
    dysample_gather4<<<(unsigned)(quads / 256), 256, 0, stream>>>(
        x, IX, IY, (float4*)d_out);
}
